// HybridGNN_22162031247392
// MI455X (gfx1250) — matrix operand order repaired and checked by reading, not compiled, not run on hardware
//
#include <hip/hip_runtime.h>
#include <hip/hip_bf16.h>

// ---------------- problem constants (from reference) ----------------
#define DIM       128     // node feature dim
#define RDK       200     // rdkit feature dim
#define HID       512     // mlp hidden
#define NAF       9       // num atom feats
#define AVOC      64      // atom vocab
#define NBF       3       // num bond feats
#define BVOC      16      // bond vocab
#define KPAD_MLP1 352     // 328 padded up to multiple of 32

typedef __bf16 bf16_t;
typedef __attribute__((ext_vector_type(16))) __bf16 v16bf;
typedef __attribute__((ext_vector_type(8)))  __bf16 v8bf;
typedef __attribute__((ext_vector_type(8)))  float  v8f;

__device__ __forceinline__ unsigned short f2bf(float f) {
  // round-to-nearest-even f32 -> bf16
  unsigned int u = __float_as_uint(f);
  unsigned int r = u + 0x7FFFu + ((u >> 16) & 1u);
  return (unsigned short)(r >> 16);
}

// ---------------- encoders ----------------
__global__ void atom_encode_kernel(const int* __restrict__ x_feat,
                                   const float* __restrict__ atom_emb,
                                   float* __restrict__ x, int N) {
  int t = blockIdx.x * blockDim.x + threadIdx.x;
  int n = t >> 7, d = t & (DIM - 1);
  if (n >= N) return;
  float s = 0.f;
#pragma unroll
  for (int f = 0; f < NAF; ++f) {
    int c = x_feat[n * NAF + f];
    s += atom_emb[((size_t)(f * AVOC + c)) * DIM + d];
  }
  x[(size_t)n * DIM + d] = s;
}

// ---------------- fused bond-encode + message + scatter-add ----------------
// agg[dst] += relu(x[src] + sum_f bond_emb[f, attr[e,f]])
__global__ void gine_scatter_kernel(const float* __restrict__ x,
                                    const int* __restrict__ edge_index,
                                    const int* __restrict__ bond_feat,
                                    const float* __restrict__ bond_emb,
                                    float* __restrict__ agg, int E) {
  int t = blockIdx.x * blockDim.x + threadIdx.x;
  int e = t >> 5;                       // one wave (32 lanes) per edge
  if (e >= E) return;
  int lane = t & 31;
  int d0 = lane << 2;                   // 4 floats per lane = 128 dims
  int src = edge_index[e];
  int dst = edge_index[E + e];
  int c0 = bond_feat[e * NBF + 0];
  int c1 = bond_feat[e * NBF + 1];
  int c2 = bond_feat[e * NBF + 2];
  float4 xs = *(const float4*)(x + (size_t)src * DIM + d0);
  float4 b0 = *(const float4*)(bond_emb + (size_t)(0 * BVOC + c0) * DIM + d0);
  float4 b1 = *(const float4*)(bond_emb + (size_t)(1 * BVOC + c1) * DIM + d0);
  float4 b2 = *(const float4*)(bond_emb + (size_t)(2 * BVOC + c2) * DIM + d0);
  float4 m;
  m.x = fmaxf(xs.x + b0.x + b1.x + b2.x, 0.f);
  m.y = fmaxf(xs.y + b0.y + b1.y + b2.y, 0.f);
  m.z = fmaxf(xs.z + b0.z + b1.z + b2.z, 0.f);
  m.w = fmaxf(xs.w + b0.w + b1.w + b2.w, 0.f);
  float* ap = agg + (size_t)dst * DIM + d0;   // agg is L2-resident (25.6MB << 192MB)
  atomicAdd(ap + 0, m.x);
  atomicAdd(ap + 1, m.y);
  atomicAdd(ap + 2, m.z);
  atomicAdd(ap + 3, m.w);
}

// ---------------- h = x + agg, converted to bf16 ----------------
__global__ void add_to_bf16_kernel(const float* __restrict__ a,
                                   const float* __restrict__ b,
                                   unsigned short* __restrict__ out, size_t total) {
  size_t i = (size_t)blockIdx.x * blockDim.x + threadIdx.x;
  if (i >= total) return;
  out[i] = f2bf(a[i] + b[i]);
}

// ---------------- W[K,Ncols] f32 -> Wt[Ncols,Kpad] bf16 (zero-padded K) ----------------
__global__ void transpose_pad_bf16_kernel(const float* __restrict__ W,
                                          unsigned short* __restrict__ Wt,
                                          int K, int Kpad, int Ncols) {
  int i = blockIdx.x * blockDim.x + threadIdx.x;
  if (i >= Ncols * Kpad) return;
  int n = i / Kpad, k = i % Kpad;
  float v = (k < K) ? W[(size_t)k * Ncols + n] : 0.f;
  Wt[i] = f2bf(v);
}

// ---------------- generic WMMA GEMM: out = act(A[M,K]bf16 @ B[K,Ncols] + bias) ----------------
// Bt is pre-transposed [Ncols, K] bf16.  M % 16 == 0, Ncols % 16 == 0, K % 32 == 0.
// One wave computes a 16-row strip across all column tiles with v_wmma_f32_16x16x32_bf16.
__global__ void wmma_gemm_kernel(const unsigned short* __restrict__ A,
                                 const unsigned short* __restrict__ Bt,
                                 const float* __restrict__ bias,
                                 float* __restrict__ outF,
                                 unsigned short* __restrict__ outB,
                                 int M, int Ncols, int K, int relu) {
  int wave = (blockIdx.x * blockDim.x + threadIdx.x) >> 5;
  int rowBase = wave << 4;
  if (rowBase >= M) return;
  int lane = threadIdx.x & 31;
  int half = lane >> 4;           // which 16-lane half
  int r    = lane & 15;
  const unsigned short* Arow = A + (size_t)(rowBase + r) * K;
  int nTiles  = Ncols >> 4;
  int kChunks = K >> 5;
  for (int ct = 0; ct < nTiles; ++ct) {
    int col = (ct << 4) + r;
    const unsigned short* Bcol = Bt + (size_t)col * K;
    v8f acc = {0.f, 0.f, 0.f, 0.f, 0.f, 0.f, 0.f, 0.f};
    for (int kc = 0; kc < kChunks; ++kc) {
      int k0 = kc << 5;
      // 16-bit A 16x32 ISA layout: lanes 0-15 hold K 0-7 (elems 0-7) and 16-23 (elems 8-15);
      // lanes 16-31 hold K 8-15 and 24-31.
      v8bf alo = *(const v8bf*)(Arow + k0 + half * 8);
      v8bf ahi = *(const v8bf*)(Arow + k0 + 16 + half * 8);
      v16bf av;
#pragma unroll
      for (int j = 0; j < 8; ++j) { av[j] = alo[j]; av[j + 8] = ahi[j]; }
      // 16-bit B 32x16 layout: lane = column, same K order as A (lanes 0-15 hold K 0-7, 16-23;
      // lanes 16-31 hold K 8-15, 24-31) -> two 16B loads from the transposed weight row.
      v8bf blo = *(const v8bf*)(Bcol + k0 + half * 8);
      v8bf bhi = *(const v8bf*)(Bcol + k0 + 16 + half * 8);
      v16bf bv;
#pragma unroll
      for (int j = 0; j < 8; ++j) { bv[j] = blo[j]; bv[j + 8] = bhi[j]; }
      acc = __builtin_amdgcn_wmma_f32_16x16x32_bf16(false, av, false, bv,
                                                    (short)0, acc, false, false);
    }
    float bc = bias ? bias[col] : 0.f;
#pragma unroll
    for (int v = 0; v < 8; ++v) {
      float val = acc[v] + bc;
      if (relu) val = fmaxf(val, 0.f);
      int rowOut = rowBase + half * 8 + v;   // C/D layout: VGPR v -> row (half*8+v), lane&15 -> col
      size_t o = (size_t)rowOut * Ncols + col;
      if (outF) outF[o] = val;
      if (outB) outB[o] = f2bf(val);
    }
  }
}

// ---------------- mean pool: sums/counts via L2 atomics ----------------
__global__ void pool_sum_kernel(const float* __restrict__ x, const int* __restrict__ batch,
                                float* __restrict__ sums, float* __restrict__ counts, int N) {
  int t = blockIdx.x * blockDim.x + threadIdx.x;
  int n = t >> 7, d = t & (DIM - 1);
  if (n >= N) return;
  int g = batch[n];
  atomicAdd(&sums[(size_t)g * DIM + d], x[(size_t)n * DIM + d]);
  if (d == 0) atomicAdd(&counts[g], 1.0f);
}

// ---------------- hcat[G, 352] bf16 = [pooled(128) | rdkit(200) | zero-pad(24)] ----------------
__global__ void build_hcat_kernel(const float* __restrict__ sums, const float* __restrict__ counts,
                                  const float* __restrict__ rdkit,
                                  unsigned short* __restrict__ hcat, int G) {
  int t = blockIdx.x * blockDim.x + threadIdx.x;
  int g = t / KPAD_MLP1, d = t % KPAD_MLP1;
  if (g >= G) return;
  float v;
  if (d < DIM)            v = sums[(size_t)g * DIM + d] / fmaxf(counts[g], 1.0f);
  else if (d < DIM + RDK) v = rdkit[(size_t)g * RDK + (d - DIM)];
  else                    v = 0.f;
  hcat[t] = f2bf(v);
}

// ---------------- final [G,256] @ [256,1] dot ----------------
__global__ void final_dot_kernel(const float* __restrict__ h2, const float* __restrict__ w3,
                                 const float* __restrict__ b3, float* __restrict__ out, int G) {
  int g = (blockIdx.x * blockDim.x + threadIdx.x) >> 5;
  if (g >= G) return;
  int lane = threadIdx.x & 31;
  float s = 0.f;
  for (int k = lane; k < HID / 2; k += 32) s += h2[(size_t)g * (HID / 2) + k] * w3[k];
#pragma unroll
  for (int off = 16; off > 0; off >>= 1) s += __shfl_down(s, off, 32);
  if (lane == 0) out[g] = s + b3[0];
}

// ---------------- host launcher ----------------
extern "C" void kernel_launch(void* const* d_in, const int* in_sizes, int n_in,
                              void* d_out, int out_size, void* d_ws, size_t ws_size,
                              hipStream_t stream) {
  const int*   x_feat   = (const int*)  d_in[0];
  const int*   edge_idx = (const int*)  d_in[1];
  const int*   bond_ft  = (const int*)  d_in[2];
  const int*   batch    = (const int*)  d_in[3];
  const float* rdkit    = (const float*)d_in[4];
  const float* atom_emb = (const float*)d_in[5];
  const float* bond_emb = (const float*)d_in[6];
  const float* c1w1 = (const float*)d_in[7];  const float* c1b1 = (const float*)d_in[8];
  const float* c1w2 = (const float*)d_in[9];  const float* c1b2 = (const float*)d_in[10];
  const float* c2w1 = (const float*)d_in[11]; const float* c2b1 = (const float*)d_in[12];
  const float* c2w2 = (const float*)d_in[13]; const float* c2b2 = (const float*)d_in[14];
  const float* mw1  = (const float*)d_in[15]; const float* mb1  = (const float*)d_in[16];
  const float* mw2  = (const float*)d_in[17]; const float* mb2  = (const float*)d_in[18];
  const float* mw3  = (const float*)d_in[19]; const float* mb3  = (const float*)d_in[20];

  const int N = in_sizes[0] / NAF;
  const int E = in_sizes[1] / 2;
  const int G = in_sizes[4] / RDK;

  // ---- workspace carve-out ----
  char* ws = (char*)d_ws; size_t off = 0;
  auto alloc = [&](size_t bytes) -> void* {
    off = (off + 255) & ~(size_t)255;
    void* p = ws + off; off += bytes; return p;
  };
  float*          x     = (float*)         alloc((size_t)N * DIM * 4);
  float*          agg   = (float*)         alloc((size_t)N * DIM * 4);
  unsigned short* hb    = (unsigned short*)alloc((size_t)N * DIM * 2);
  unsigned short* h2b   = (unsigned short*)alloc((size_t)N * DIM * 2);
  unsigned short* wtc[4];
  for (int i = 0; i < 4; ++i) wtc[i] = (unsigned short*)alloc((size_t)DIM * DIM * 2);
  float*          sums  = (float*)         alloc((size_t)G * DIM * 4);
  float*          cnts  = (float*)         alloc((size_t)G * 4);
  unsigned short* hcat  = (unsigned short*)alloc((size_t)G * KPAD_MLP1 * 2);
  unsigned short* wtm1  = (unsigned short*)alloc((size_t)HID * KPAD_MLP1 * 2);
  unsigned short* wtm2  = (unsigned short*)alloc((size_t)(HID / 2) * HID * 2);
  unsigned short* h1b   = (unsigned short*)alloc((size_t)G * HID * 2);
  float*          h2f   = (float*)         alloc((size_t)G * (HID / 2) * 4);
  (void)ws_size; (void)n_in; (void)out_size;

  const int TB = 256;
  // ---- weight transposes (bf16, zero-padded K) ----
  const float* cw[4] = {c1w1, c1w2, c2w1, c2w2};
  for (int i = 0; i < 4; ++i)
    transpose_pad_bf16_kernel<<<(DIM * DIM + TB - 1) / TB, TB, 0, stream>>>(cw[i], wtc[i], DIM, DIM, DIM);
  transpose_pad_bf16_kernel<<<(HID * KPAD_MLP1 + TB - 1) / TB, TB, 0, stream>>>(mw1, wtm1, DIM + RDK, KPAD_MLP1, HID);
  transpose_pad_bf16_kernel<<<((HID / 2) * HID + TB - 1) / TB, TB, 0, stream>>>(mw2, wtm2, HID, HID, HID / 2);

  // ---- atom encoder ----
  atom_encode_kernel<<<((size_t)N * DIM + TB - 1) / TB, TB, 0, stream>>>(x_feat, atom_emb, x, N);

  // ---- two GINE convs ----
  size_t nd = (size_t)N * DIM;
  int gemmBlocks = (int)(((size_t)(N / 16) + 3) / 4);   // 4 waves/block, 1 strip each
  for (int conv = 0; conv < 2; ++conv) {
    const unsigned short* w1t = wtc[conv * 2 + 0];
    const unsigned short* w2t = wtc[conv * 2 + 1];
    const float* b1 = (conv == 0) ? c1b1 : c2b1;
    const float* b2 = (conv == 0) ? c1b2 : c2b2;
    hipMemsetAsync(agg, 0, nd * 4, stream);
    gine_scatter_kernel<<<((size_t)E * 32 + TB - 1) / TB, TB, 0, stream>>>(
        x, edge_idx, bond_ft, bond_emb, agg, E);
    add_to_bf16_kernel<<<(nd + TB - 1) / TB, TB, 0, stream>>>(x, agg, hb, nd);
    // h2 = relu(h @ W1 + b1)  -> bf16
    wmma_gemm_kernel<<<gemmBlocks, 128, 0, stream>>>(hb, w1t, b1, (float*)nullptr, h2b,
                                                     N, DIM, DIM, 1);
    // x  = h2 @ W2 + b2       -> f32 (in place over dead x)
    wmma_gemm_kernel<<<gemmBlocks, 128, 0, stream>>>(h2b, w2t, b2, x, (unsigned short*)nullptr,
                                                     N, DIM, DIM, 0);
  }

  // ---- global mean pool + concat ----
  hipMemsetAsync(sums, 0, (size_t)G * DIM * 4, stream);
  hipMemsetAsync(cnts, 0, (size_t)G * 4, stream);
  pool_sum_kernel<<<(nd + TB - 1) / TB, TB, 0, stream>>>(x, batch, sums, cnts, N);
  build_hcat_kernel<<<((size_t)G * KPAD_MLP1 + TB - 1) / TB, TB, 0, stream>>>(
      sums, cnts, rdkit, hcat, G);

  // ---- MLP head ----
  int gemmBlocksG = (G / 16 + 3) / 4;
  wmma_gemm_kernel<<<gemmBlocksG, 128, 0, stream>>>(hcat, wtm1, mb1, (float*)nullptr, h1b,
                                                    G, HID, KPAD_MLP1, 1);
  wmma_gemm_kernel<<<gemmBlocksG, 128, 0, stream>>>(h1b, wtm2, mb2, h2f, (unsigned short*)nullptr,
                                                    G, HID / 2, HID, 1);
  final_dot_kernel<<<(G * 32 + 127) / 128, 128, 0, stream>>>(h2f, mw3, mb3, (float*)d_out, G);
}
